// MultiHeadAttention_6012954214564
// MI455X (gfx1250) — matrix operand order repaired and checked by reading, not compiled, not run on hardware
//
#include <hip/hip_runtime.h>
#include <hip/hip_fp16.h>

// ---------------------------------------------------------------------------
// MHA forward for MI455X (gfx1250, wave32, WMMA).
//   B=2, S=2048, D=1024, H=16, DH=64. ~70 GFLOP over ~80MB -> compute bound:
//   all GEMMs run on v_wmma_f32_16x16x32_f16; attention is streamed (online
//   softmax) so S^2 score matrices never touch HBM and f16 K/V (~16MB) stay
//   resident in the 192MB L2.
// Round 2: raise WMMA density — GEMM wave tile 32x64 (B-fragment reuse x2),
//   flash key chunk 64 (one softmax rescale per 16 WMMAs instead of 8).
// ---------------------------------------------------------------------------

typedef _Float16 f16;
typedef __attribute__((ext_vector_type(16))) _Float16 v16h;
typedef __attribute__((ext_vector_type(8)))  _Float16 v8h;
typedef __attribute__((ext_vector_type(4)))  _Float16 v4h;
typedef __attribute__((ext_vector_type(8)))  float    v8f;
typedef __attribute__((ext_vector_type(4)))  float    v4f;

constexpr int Bc  = 2;
constexpr int Sc  = 2048;
constexpr int Dc  = 1024;
constexpr int Hc  = 16;
constexpr int DHc = 64;
constexpr float SCALEc = 0.125f;     // DH^-0.5
constexpr float NEGBIG = -1.0e30f;   // finite "-inf" sentinel (keeps exp() NaN-free)

// ---------------------------------------------------------------------------
// f32 -> f16 elementwise convert (vectorized by 4)
// ---------------------------------------------------------------------------
__global__ void cvt_f32_to_f16(const float* __restrict__ src,
                               f16* __restrict__ dst, int n4) {
  int i = blockIdx.x * blockDim.x + threadIdx.x;
  if (i < n4) {
    v4f v = ((const v4f*)src)[i];
    v4h o;
#pragma unroll
    for (int j = 0; j < 4; ++j) o[j] = (f16)v[j];
    ((v4h*)dst)[i] = o;
  }
}

// ---------------------------------------------------------------------------
// WMMA GEMM: C[M,N] = A[M,K](f16) @ W[N,K]^T(f16) + bias(f32)
// Block = 128 threads (4 waves). Wave tile = 32(M) x 64(N): 2 M-subtiles x
// 4 N-subtiles = 8 accumulators; each B fragment feeds 2 WMMAs. Block tile =
// 128 x 64, K-step 32.
// MODE 0: f16 out, [B,H,S,DH]   (q / k heads)
// MODE 1: f16 out, [B,H,DH,S]   (v transposed: keys contiguous for P@V frags)
// MODE 2: f32 out, row-major [M,N]  (final projection -> d_out)
// ---------------------------------------------------------------------------
template <int MODE>
__global__ __launch_bounds__(128)
void gemm_wmma_kernel(const f16* __restrict__ A, const f16* __restrict__ W,
                      const float* __restrict__ bias, void* __restrict__ out,
                      int M, int N, int K) {
  const int lane  = threadIdx.x & 31;
  const int wave  = threadIdx.x >> 5;
  const int col16 = lane & 15;
  const int half  = lane >> 4;            // 0: lanes 0-15, 1: lanes 16-31
  const int nbase = blockIdx.x * 64;
  const int mbase = blockIdx.y * 128 + wave * 32;

  v8f acc[2][4] = {};

  // A fragment (16x32 f16): lane<16 holds row=lane, K {k0..k0+15};
  // lane>=16 holds row=lane-16, K {k0+16..k0+31}.
  const f16* aptr0 = A + (size_t)(mbase + col16) * K + (half ? 16 : 0);
  const f16* aptr1 = A + (size_t)(mbase + 16 + col16) * K + (half ? 16 : 0);

  for (int k0 = 0; k0 < K; k0 += 32) {
    v16h afrag[2];
#pragma unroll
    for (int mt = 0; mt < 2; ++mt) {
      const f16* ap = (mt ? aptr1 : aptr0) + k0;
      v8h lo = *(const v8h*)ap;
      v8h hi = *(const v8h*)(ap + 8);
#pragma unroll
      for (int i = 0; i < 8; ++i) { afrag[mt][i] = lo[i]; afrag[mt][i + 8] = hi[i]; }
    }
    if (k0 + 64 < K) {                                   // global_prefetch_b8
      __builtin_prefetch(aptr0 + k0 + 64, 0, 0);
      __builtin_prefetch(aptr1 + k0 + 64, 0, 0);
    }

#pragma unroll
    for (int nt = 0; nt < 4; ++nt) {
      // B fragment (32x16): lane<16 holds col=n, K {k0..k0+15}; lane>=16 K
      // {k0+16..k0+31}. B[k][n] = W[n][k] -> row n of W contiguous in K (32B).
      const int bcol = nbase + nt * 16 + col16;
      const f16* bptr = W + (size_t)bcol * K + k0 + (half ? 16 : 0);
      v16h bfrag = *(const v16h*)bptr;
      acc[0][nt] = __builtin_amdgcn_wmma_f32_16x16x32_f16(
          false, afrag[0], false, bfrag, (short)0, acc[0][nt], false, false);
      acc[1][nt] = __builtin_amdgcn_wmma_f32_16x16x32_f16(
          false, afrag[1], false, bfrag, (short)0, acc[1][nt], false, false);
    }
  }

  // Epilogue. C layout: VGPR g -> row (g + 8*half), col = lane&15 within tile.
#pragma unroll
  for (int mt = 0; mt < 2; ++mt) {
#pragma unroll
    for (int nt = 0; nt < 4; ++nt) {
#pragma unroll
      for (int g = 0; g < 8; ++g) {
        const int r = mbase + mt * 16 + g + 8 * half;   // global row (m)
        const int c = nbase + nt * 16 + col16;          // global col (n)
        const float v = acc[mt][nt][g] + bias[c];
        if constexpr (MODE == 2) {
          ((float*)out)[(size_t)r * N + c] = v;
        } else {
          const int b = r >> 11, s = r & (Sc - 1);      // M = B*S, S = 2048
          const int h = c >> 6, d = c & (DHc - 1);      // N = D = H*DH
          f16* o16 = (f16*)out;
          if constexpr (MODE == 0)
            o16[(((size_t)(b * Hc + h)) * Sc + s) * DHc + d] = (f16)v;  // [B,H,S,DH]
          else
            o16[(((size_t)(b * Hc + h)) * DHc + d) * Sc + s] = (f16)v;  // [B,H,DH,S]
        }
      }
    }
  }
}

// ---------------------------------------------------------------------------
// Flash attention (causal + padding mask), one (b,h) x 64-query tile per block.
// 4 waves; wave owns 16 query rows x DH=64 output. Keys streamed in chunks of
// 64: 8 score WMMAs (4 subtiles x K=64), ONE online-softmax rescale, P tile
// transposed C-layout -> A-fragments via per-wave LDS, 8 P@V WMMAs.
// ---------------------------------------------------------------------------
__global__ __launch_bounds__(128)
void flash_attn_kernel(const f16* __restrict__ qh, const f16* __restrict__ kh,
                       const f16* __restrict__ vt, const int* __restrict__ mask,
                       f16* __restrict__ ctx) {
  __shared__ __align__(64) f16 ptile[4][16 * 64];   // per-wave 16x64 P tile (8KB)

  const int lane  = threadIdx.x & 31;
  const int wave  = threadIdx.x >> 5;
  const int col16 = lane & 15;
  const int half  = lane >> 4;
  const int bh    = blockIdx.y;
  const int b     = bh >> 4;          // H = 16
  const int h     = bh & 15;
  const int qtile = blockIdx.x;
  const int qbase = qtile * 64 + wave * 16;

  const f16* qptr = qh + (size_t)bh * Sc * DHc;     // [S, DH]
  const f16* kptr = kh + (size_t)bh * Sc * DHc;     // [S, DH]
  const f16* vptr = vt + (size_t)bh * DHc * Sc;     // [DH, S] (transposed)
  const int* mrow = mask + b * Sc;

  // Q A-fragments for both K-steps of DH=64.
  const int qrow = qbase + col16;
  v16h aq[2];
#pragma unroll
  for (int ks = 0; ks < 2; ++ks) {
    const f16* ap = qptr + (size_t)qrow * DHc + ks * 32 + (half ? 16 : 0);
    v8h lo = *(const v8h*)ap;
    v8h hi = *(const v8h*)(ap + 8);
#pragma unroll
    for (int i = 0; i < 8; ++i) { aq[ks][i] = lo[i]; aq[ks][i + 8] = hi[i]; }
  }

  v8f  o[4] = {};
  float m[8], l[8];
#pragma unroll
  for (int g = 0; g < 8; ++g) { m[g] = NEGBIG; l[g] = 0.0f; }

  // Uniform chunk count across the block (keys up to qtile*64+63) so
  // __syncthreads below is legal; over-covered rows just get p==0.
  const int nchunks = qtile + 1;

  for (int kc = 0; kc < nchunks; ++kc) {
    const int kb0 = kc * 64;

    // ---- scores: four 16x16 subtiles over keys [kb0, kb0+64) ----
    v8f sv[4];
#pragma unroll
    for (int sub = 0; sub < 4; ++sub) {
      const int key = kb0 + sub * 16 + col16;
      const f16* kp = kptr + (size_t)key * DHc + (half ? 16 : 0);
      v16h bk0 = *(const v16h*)kp;          // dk 0..31  (B frag: col=key)
      v16h bk1 = *(const v16h*)(kp + 32);   // dk 32..63
      v8f s = {};
      s = __builtin_amdgcn_wmma_f32_16x16x32_f16(false, aq[0], false, bk0,
                                                 (short)0, s, false, false);
      s = __builtin_amdgcn_wmma_f32_16x16x32_f16(false, aq[1], false, bk1,
                                                 (short)0, s, false, false);
      const int mv = mrow[key];
#pragma unroll
      for (int g = 0; g < 8; ++g) {
        const int row = qbase + g + 8 * half;
        float x = s[g] * SCALEc;
        if (key > row || mv == 0) x = NEGBIG;   // causal + padding mask
        s[g] = x;
      }
      sv[sub] = s;
    }

    // ---- one online-softmax rescale per 64 keys ----
    float alpha[8];
#pragma unroll
    for (int g = 0; g < 8; ++g) {
      float cm = fmaxf(fmaxf(sv[0][g], sv[1][g]), fmaxf(sv[2][g], sv[3][g]));
      cm = fmaxf(cm, __shfl_xor(cm, 1));
      cm = fmaxf(cm, __shfl_xor(cm, 2));
      cm = fmaxf(cm, __shfl_xor(cm, 4));
      cm = fmaxf(cm, __shfl_xor(cm, 8));
      const float mn = fmaxf(m[g], cm);
      alpha[g] = __expf(m[g] - mn);
      float rs = 0.0f;
#pragma unroll
      for (int sub = 0; sub < 4; ++sub) {
        const float p = __expf(sv[sub][g] - mn);
        sv[sub][g] = p;
        rs += p;
      }
      rs += __shfl_xor(rs, 1);
      rs += __shfl_xor(rs, 2);
      rs += __shfl_xor(rs, 4);
      rs += __shfl_xor(rs, 8);
      l[g] = l[g] * alpha[g] + rs;
      m[g] = mn;
    }
#pragma unroll
    for (int t = 0; t < 4; ++t)
#pragma unroll
      for (int g = 0; g < 8; ++g) o[t][g] *= alpha[g];

    // ---- P: C-layout -> row-major 16x64 in LDS ----
    f16* pt = &ptile[wave][0];
#pragma unroll
    for (int g = 0; g < 8; ++g) {
      const int r = g + 8 * half;
#pragma unroll
      for (int sub = 0; sub < 4; ++sub)
        pt[r * 64 + sub * 16 + col16] = (f16)sv[sub][g];
    }
    __syncthreads();

    // ---- O += P @ V[kb0:kb0+64, :] : 2 K-steps x 4 DH-subtiles ----
#pragma unroll
    for (int ks = 0; ks < 2; ++ks) {
      const f16* pp = pt + col16 * 64 + ks * 32 + (half ? 16 : 0);
      v8h plo = *(const v8h*)pp;
      v8h phi = *(const v8h*)(pp + 8);
      v16h pfrag;
#pragma unroll
      for (int i = 0; i < 8; ++i) { pfrag[i] = plo[i]; pfrag[i + 8] = phi[i]; }
#pragma unroll
      for (int t = 0; t < 4; ++t) {
        const int dcol = t * 16 + col16;
        const f16* vp = vptr + (size_t)dcol * Sc + kb0 + ks * 32 + (half ? 16 : 0);
        v16h bv = *(const v16h*)vp;
        o[t] = __builtin_amdgcn_wmma_f32_16x16x32_f16(false, pfrag, false, bv,
                                                      (short)0, o[t], false, false);
      }
    }
    __syncthreads();   // protect LDS tile from next iteration's stores
  }

  // ---- normalize and scatter to ctx [B, S, D] (heads re-interleaved) ----
#pragma unroll
  for (int g = 0; g < 8; ++g) l[g] = 1.0f / l[g];
#pragma unroll
  for (int t = 0; t < 4; ++t) {
#pragma unroll
    for (int g = 0; g < 8; ++g) {
      const int r = qbase + g + 8 * half;
      const int c = h * DHc + t * 16 + col16;
      ctx[((size_t)b * Sc + r) * Dc + c] = (f16)(o[t][g] * l[g]);
    }
  }
}

// ---------------------------------------------------------------------------
// Host orchestration
// ---------------------------------------------------------------------------
extern "C" void kernel_launch(void* const* d_in, const int* in_sizes, int n_in,
                              void* d_out, int out_size, void* d_ws, size_t ws_size,
                              hipStream_t stream) {
  (void)in_sizes; (void)n_in; (void)out_size; (void)ws_size;

  const float* Q   = (const float*)d_in[0];
  const float* K   = (const float*)d_in[1];
  const float* V   = (const float*)d_in[2];
  const int*   msk = (const int*)d_in[3];
  const float* Wq  = (const float*)d_in[4];
  const float* bq  = (const float*)d_in[5];
  const float* Wk  = (const float*)d_in[6];
  const float* bk  = (const float*)d_in[7];
  const float* Wv  = (const float*)d_in[8];
  const float* bv  = (const float*)d_in[9];
  const float* Wo  = (const float*)d_in[10];
  const float* bo  = (const float*)d_in[11];
  float* out = (float*)d_out;

  const size_t nBSD = (size_t)Bc * Sc * Dc;   // 4M elems
  const size_t nDD  = (size_t)Dc * Dc;        // 1M elems

  f16* w = (f16*)d_ws;                        // total: 32M f16 = 64 MB
  f16* Q16  = w; w += nBSD;
  f16* K16  = w; w += nBSD;
  f16* V16  = w; w += nBSD;
  f16* Wq16 = w; w += nDD;
  f16* Wk16 = w; w += nDD;
  f16* Wv16 = w; w += nDD;
  f16* Wo16 = w; w += nDD;
  f16* qh   = w; w += nBSD;                   // [B,H,S,DH]
  f16* kh   = w; w += nBSD;                   // [B,H,S,DH]
  f16* vt   = w; w += nBSD;                   // [B,H,DH,S]
  f16* ctx  = w; w += nBSD;                   // [B,S,D]

  auto cvt = [&](const float* s, f16* d, size_t n) {
    int n4 = (int)(n / 4);
    hipLaunchKernelGGL(cvt_f32_to_f16, dim3((n4 + 255) / 256), dim3(256), 0,
                       stream, s, d, n4);
  };
  cvt(Q, Q16, nBSD);  cvt(K, K16, nBSD);  cvt(V, V16, nBSD);
  cvt(Wq, Wq16, nDD); cvt(Wk, Wk16, nDD); cvt(Wv, Wv16, nDD); cvt(Wo, Wo16, nDD);

  const dim3 gGrid(Dc / 64, (Bc * Sc) / 128);  // (16, 32)
  const dim3 gBlk(128);
  hipLaunchKernelGGL((gemm_wmma_kernel<0>), gGrid, gBlk, 0, stream,
                     Q16, Wq16, bq, (void*)qh, Bc * Sc, Dc, Dc);
  hipLaunchKernelGGL((gemm_wmma_kernel<0>), gGrid, gBlk, 0, stream,
                     K16, Wk16, bk, (void*)kh, Bc * Sc, Dc, Dc);
  hipLaunchKernelGGL((gemm_wmma_kernel<1>), gGrid, gBlk, 0, stream,
                     V16, Wv16, bv, (void*)vt, Bc * Sc, Dc, Dc);

  hipLaunchKernelGGL(flash_attn_kernel, dim3(Sc / 64, Bc * Hc), dim3(128), 0,
                     stream, qh, kh, vt, msk, ctx);

  hipLaunchKernelGGL((gemm_wmma_kernel<2>), gGrid, gBlk, 0, stream,
                     ctx, Wo16, bo, (void*)out, Bc * Sc, Dc, Dc);
}
